// TokenLearner_29721173688666
// MI455X (gfx1250) — matrix operand order repaired and checked by reading, not compiled, not run on hardware
//
#include <hip/hip_runtime.h>
#include <hip/hip_bf16.h>

typedef __attribute__((ext_vector_type(16))) __bf16 v16bf;
typedef __attribute__((ext_vector_type(8)))  __bf16 v8bf;
typedef __attribute__((ext_vector_type(8)))  float  v8f;

union Frag16 { v16bf v; v8bf h[2]; };

// Problem constants
constexpr int Bn = 8, Cn = 256, Dn = 16, Hn = 32, Wn = 32, Sn = 8;
constexpr int Nn = Dn * Hn * Wn;                 // 16384
constexpr int KSTEPS = 8 * 27;                   // 8 c-chunks * 27 taps = 216
constexpr size_t WPACK_ELEMS = (size_t)KSTEPS * 32 * 16;   // 110592 bf16
constexpr size_t WPACK_BYTES = WPACK_ELEMS * 2;            // 221184 B (256-aligned)
constexpr size_t SEL_ELEMS   = (size_t)Bn * 16 * Nn;       // padded S=16

// ---------------------------------------------------------------------------
// Kernel 0: pack conv weights (f32, S x C x 3x3x3) into per-lane WMMA
// A-fragment order: wpack[(kstep*32 + lane)*16 + v*2 + e], kstep = cc*27+tap.
// Rows s>=8 are zero (S padded 8 -> 16).
// ---------------------------------------------------------------------------
__global__ __launch_bounds__(256) void pack_w_kernel(
    const float* __restrict__ Wc, __bf16* __restrict__ wpack) {
  int g = blockIdx.x * 256 + threadIdx.x;        // 0..6911 (27 blocks)
  if (g >= KSTEPS * 32) return;
  int kstep = g >> 5, lane = g & 31;
  int cc = kstep / 27, tap = kstep % 27;
  int kd = tap / 9, kh = (tap / 3) % 3, kw = tap % 3;
  int s = lane & 15;
  __bf16* dst = wpack + (size_t)g * 16;
#pragma unroll
  for (int v = 0; v < 8; ++v) {
#pragma unroll
    for (int e = 0; e < 2; ++e) {
      // K index within the 32-wide k-step for (lane, VGPR v, half e)
      int K = 2 * (v & 3) + ((v >= 4) ? 16 : 0) + ((lane >= 16) ? 8 : 0) + e;
      int c = cc * 32 + K;
      float val = 0.0f;
      if (s < Sn)
        val = Wc[(((size_t)(s * Cn + c) * 3 + kd) * 3 + kh) * 3 + kw];
      dst[v * 2 + e] = (__bf16)val;
    }
  }
}

// ---------------------------------------------------------------------------
// Kernel 1: zero the f32 output (needed for split-K atomics).
// ---------------------------------------------------------------------------
__global__ void zero_out_kernel(float* __restrict__ out, int n) {
  int i = blockIdx.x * 256 + threadIdx.x;
  if (i < n) out[i] = 0.0f;
}

// ---------------------------------------------------------------------------
// Kernel 2: conv3d(C->S, k=3, pad=1) + sigmoid, implicit GEMM with
// v_wmma_f32_16x16x32_bf16.  D[s,m] = sum_k Wpack[s,k] * patch[k,m].
// Workgroup: 8 waves, tile = one (b,d), 4 h-rows, full W (128 points).
// x staged per 32-channel chunk into LDS, channel-fastest, double-buffered.
// ---------------------------------------------------------------------------
constexpr int LDS_DZ = 3, LDS_HY = 6, LDS_WZ = 34, LDS_CP = 40; // 80B rows
constexpr int LDS_PER_BUF = LDS_DZ * LDS_HY * LDS_WZ * LDS_CP;  // 24480 bf16

__global__ __launch_bounds__(256) void conv_sigmoid_kernel(
    const float* __restrict__ x, const float* __restrict__ bias,
    const __bf16* __restrict__ wpack, __bf16* __restrict__ sel) {
  __shared__ __align__(16) __bf16 lds[2][LDS_PER_BUF];

  const int tid = threadIdx.x;
  const int t = blockIdx.x;                 // 0..1023 = B * D * (H/4)
  const int b = t >> 7;                     // /128
  const int rem = t & 127;
  const int d = rem >> 3;                   // 0..15
  const int h0 = (rem & 7) << 2;            // 0,4,...,28

  // Zero whole LDS once: w-halo columns (wz==0, wz==33) stay zero forever.
  for (int e = tid; e < 2 * LDS_PER_BUF; e += 256)
    ((__bf16*)lds)[e] = (__bf16)0.0f;
  __syncthreads();

  auto stage = [&](int cc, int buf) {
    const int c0 = cc * 32;
    // e = ((cl*3 + dz)*6 + hy)*32 + w : w fastest -> coalesced global reads
    for (int e = tid; e < 32 * 3 * 6 * 32; e += 256) {
      int w = e & 31;
      int r = e >> 5;
      int hy = r % 6; r /= 6;
      int dz = r % 3;
      int cl = r / 3;
      int dd = d + dz - 1, hh = h0 + hy - 1;
      float v = 0.0f;
      if (dd >= 0 && dd < Dn && hh >= 0 && hh < Hn)
        v = x[(((size_t)(b * Cn + c0 + cl) * Dn + dd) * Hn + hh) * Wn + w];
      lds[buf][((dz * LDS_HY + hy) * LDS_WZ + (w + 1)) * LDS_CP + cl] = (__bf16)v;
    }
  };

  const int wave = tid >> 5, lane = tid & 31;
  const int hrow = wave >> 1;                       // 0..3
  const int wcol = ((wave & 1) << 4) + (lane & 15); // 0..31
  const int clo = (lane < 16) ? 0 : 8;              // K-half -> c_local base

  v8f acc = {};

  stage(0, 0);
  __syncthreads();

  for (int cc = 0; cc < 8; ++cc) {
    const int buf = cc & 1;
    if (cc + 1 < 8) stage(cc + 1, buf ^ 1);
    const __bf16* base = lds[buf];
#pragma unroll
    for (int tap = 0; tap < 27; ++tap) {
      const int kd = tap / 9, kh = (tap / 3) % 3, kw = tap % 3;
      Frag16 a, bp;
      const __bf16* ap = wpack + ((size_t)((cc * 27 + tap) * 32 + lane)) * 16;
      a.h[0] = *(const v8bf*)(ap);
      a.h[1] = *(const v8bf*)(ap + 8);
      const __bf16* lp = base +
          ((kd * LDS_HY + hrow + kh) * LDS_WZ + (wcol + kw)) * LDS_CP + clo;
      bp.h[0] = *(const v8bf*)(lp);
      bp.h[1] = *(const v8bf*)(lp + 16);
      acc = __builtin_amdgcn_wmma_f32_16x16x32_bf16(
          false, a.v, false, bp.v, (short)0, acc, false, false);
    }
    __syncthreads();
  }

  // Epilogue: D layout -> lane<16 holds rows s=v, lane>=16 rows s=v+8 (pad).
  const int n_m = (d * Hn + (h0 + hrow)) * Wn + wcol;
#pragma unroll
  for (int v = 0; v < 8; ++v) {
    const int s = v + ((lane < 16) ? 0 : 8);
    float val = 0.0f;
    if (lane < 16) {
      float tc = acc[v] + bias[s];
      val = 1.0f / (1.0f + __expf(-tc));
    }
    sel[(size_t)(b * 16 + s) * Nn + n_m] = (__bf16)val;     // rows 8..15 = 0
  }
}

// ---------------------------------------------------------------------------
// Kernel 3: out[b,s,c] = (1/N) * sum_n sel[b,s,n] * x[b,c,n]
// GEMM per b: (16(S-pad) x N) * (N x 16-c-block), split-K over n, f32 atomics.
// ---------------------------------------------------------------------------
__global__ __launch_bounds__(256) void contract_kernel(
    const float* __restrict__ x, const __bf16* __restrict__ sel,
    float* __restrict__ out) {
  const int wt = blockIdx.x * 8 + (threadIdx.x >> 5);  // 0..8191 wave tasks
  const int lane = threadIdx.x & 31;
  const int b = wt >> 10;                   // /1024 (16 cblk * 64 ksplit)
  const int r = wt & 1023;
  const int cblk = r >> 6;
  const int ks = r & 63;
  const int c0 = cblk * 16;
  const int nbase = ks * 256;
  const int row = lane & 15;                // A row (s) and B column (c)
  const int half = lane >> 4;               // K-half select

  const __bf16* selp = sel + (size_t)(b * 16 + row) * Nn + nbase + half * 16;
  const float* xp = x + ((size_t)b * Cn + (c0 + row)) * Nn + nbase + half * 16;

  v8f acc = {};
#pragma unroll
  for (int t8 = 0; t8 < 8; ++t8) {
    Frag16 a, bp;
    const __bf16* ap = selp + t8 * 32;
    a.h[0] = *(const v8bf*)(ap);            // K = half*16 + 0..7
    a.h[1] = *(const v8bf*)(ap + 8);        // K = half*16 + 8..15
    const float* fp = xp + t8 * 32;
    float4 f0 = *(const float4*)(fp);
    float4 f1 = *(const float4*)(fp + 4);
    float4 f2 = *(const float4*)(fp + 8);
    float4 f3 = *(const float4*)(fp + 12);
    bp.v[0]  = (__bf16)f0.x; bp.v[1]  = (__bf16)f0.y;
    bp.v[2]  = (__bf16)f0.z; bp.v[3]  = (__bf16)f0.w;
    bp.v[4]  = (__bf16)f1.x; bp.v[5]  = (__bf16)f1.y;
    bp.v[6]  = (__bf16)f1.z; bp.v[7]  = (__bf16)f1.w;
    bp.v[8]  = (__bf16)f2.x; bp.v[9]  = (__bf16)f2.y;
    bp.v[10] = (__bf16)f2.z; bp.v[11] = (__bf16)f2.w;
    bp.v[12] = (__bf16)f3.x; bp.v[13] = (__bf16)f3.y;
    bp.v[14] = (__bf16)f3.z; bp.v[15] = (__bf16)f3.w;
    acc = __builtin_amdgcn_wmma_f32_16x16x32_bf16(
        false, a.v, false, bp.v, (short)0, acc, false, false);
  }

  const float scale = 1.0f / (float)Nn;
  if (lane < 16) {                          // rows s = v (0..7) only
#pragma unroll
    for (int v = 0; v < 8; ++v)
      atomicAdd(out + (size_t)(b * Sn + v) * Cn + (c0 + row), acc[v] * scale);
  }
}

// ---------------------------------------------------------------------------
extern "C" void kernel_launch(void* const* d_in, const int* in_sizes, int n_in,
                              void* d_out, int out_size, void* d_ws, size_t ws_size,
                              hipStream_t stream) {
  const float* x  = (const float*)d_in[0];   // (8,256,16,32,32)
  const float* Wc = (const float*)d_in[1];   // (8,256,3,3,3)
  const float* bb = (const float*)d_in[2];   // (8,)
  float* out = (float*)d_out;                // (8,8,256)

  __bf16* wpack = (__bf16*)d_ws;
  __bf16* sel   = (__bf16*)((char*)d_ws + WPACK_BYTES);

  pack_w_kernel<<<27, 256, 0, stream>>>(Wc, wpack);
  zero_out_kernel<<<(out_size + 255) / 256, 256, 0, stream>>>(out, out_size);
  conv_sigmoid_kernel<<<Bn * Dn * (Hn / 4), 256, 0, stream>>>(x, bb, wpack, sel);
  contract_kernel<<<1024, 256, 0, stream>>>(x, sel, out);
}
